// Attention_42021960024366
// MI455X (gfx1250) — compile-verified, matrix operand order checked by reading, not run on hardware
//
#include <hip/hip_runtime.h>
#include <hip/hip_bf16.h>

// ============================================================================
// MI455X (gfx1250) attention block.
//   out = proj( softmax(Q Kt * sc) V ), Q/K/V from qkv GEMM.
// Strategy: bf16 WMMA (v_wmma_f32_16x16x32_bf16) everywhere, fp32 accum.
//   - GEMMs computed transposed so A and B fragments share one row-major loader
//   - GEMMs ping-pong unrolled x2: no cross-iteration register moves, no
//     WMMA->VALU WAR hazard NOPs, partial s_wait_loadcnt overlap
//   - attention computes S^T = K*Q^T so softmax is lane-local (one shfl_xor 16)
//   - online softmax runs in base-2 (log2e folded into Q scale): exp2f -> bare
//     v_exp_f32; P-pack via __builtin_convertvector -> packed v_cvt_pk_bf16_f32
//   - S^T accumulator layout == B-fragment layout of P^T (zero-shuffle feed
//     into ctx^T = V^T * P^T); V is stored pre-transposed by the QKV kernel
//   - attention stages K/V tiles via global_load_async_to_lds_b128 (ASYNCcnt),
//     double-buffered in LDS, shared by all 4 waves of the block
// ============================================================================

typedef __bf16 bf16_t;
typedef __attribute__((ext_vector_type(16))) __bf16 v16bf;
typedef __attribute__((ext_vector_type(8)))  __bf16 v8bf;
typedef __attribute__((ext_vector_type(4)))  __bf16 v4bf;
typedef __attribute__((ext_vector_type(8)))  float  v8f;
typedef __attribute__((ext_vector_type(4)))  float  v4f;

#define DIMC      768
#define NHEADS    12
#define HEADD     64
#define BATCH     8
#define SEQ       1024
#define TOKENS    (BATCH * SEQ)        // 8192
#define QKV_OUT   (3 * DIMC)           // 2304
// softmax computed in base-2: fold 64^-0.5 * log2(e) into Q
#define Q_SCALE   (0.125f * 1.44269504088896340736f)

__device__ __forceinline__ bf16_t f2bf(float f) {
  union { float f; unsigned u; } v; v.f = f;
  unsigned r = v.u + 0x7FFFu + ((v.u >> 16) & 1u);   // round-to-nearest-even
  unsigned short h = (unsigned short)(r >> 16);
  return __builtin_bit_cast(bf16_t, h);
}

// Load one 16x32 bf16 operand tile (A- or B-fragment; same lane mapping) from
// row-major memory (global -> global_load_b128, LDS -> ds_load_b128).
// lane(0..15)=row, K in {kh..kh+7} U {16+kh..16+kh+7}, kh = 8*(lane>=16).
__device__ __forceinline__ v16bf load_ab_frag(const bf16_t* __restrict__ tile,
                                              int pitch) {
  const int lane = threadIdx.x & 31;
  const int r    = lane & 15;
  const int kh   = (lane & 16) >> 1;         // 0 or 8
  const bf16_t* p = tile + r * pitch + kh;
  v8bf lo = *(const v8bf*)(p);
  v8bf hi = *(const v8bf*)(p + 16);
  v16bf f;
#pragma unroll
  for (int i = 0; i < 8; ++i) { f[i] = lo[i]; f[i + 8] = hi[i]; }
  return f;
}

__device__ __forceinline__ v8f wmma_bf16(v16bf a, v16bf b, v8f c) {
  return __builtin_amdgcn_wmma_f32_16x16x32_bf16(
      /*neg_a=*/false, a, /*neg_b=*/false, b,
      /*c_mod=*/(short)0, c, /*reuse_a=*/false, /*reuse_b=*/false);
}

// Async copy of one 64-row x 64-bf16 tile (8 KB) global -> LDS, 128 threads,
// 4x 16B chunks per thread. Tracked on ASYNCcnt.
__device__ __forceinline__ void stage_tile_async(bf16_t* lds_tile,
                                                 const bf16_t* __restrict__ gsrc,
                                                 int gpitch, int tid) {
#pragma unroll
  for (int c = 0; c < 4; ++c) {
    const int chunk = tid + c * 128;       // 0..511
    const int row   = chunk >> 3;          // 0..63
    const int col   = (chunk & 7) * 8;     // element offset, 16B aligned
    unsigned lds_addr = (unsigned)(uintptr_t)(lds_tile + row * 64 + col);
    const void* g = (const void*)(gsrc + (size_t)row * gpitch + col);
    asm volatile("global_load_async_to_lds_b128 %0, %1, off"
                 :: "v"(lds_addr), "v"(g) : "memory");
  }
}

__device__ __forceinline__ void wait_async_all() {
  asm volatile("s_wait_asynccnt 0x0" ::: "memory");
}

// ---------------------------------------------------------------------------
// fp32 -> bf16 conversion (vectorized x4)
// ---------------------------------------------------------------------------
__global__ __launch_bounds__(256)
void cvt_f32_bf16_kernel(const float* __restrict__ src,
                         bf16_t* __restrict__ dst, int n4) {
  int i = blockIdx.x * 256 + threadIdx.x;
  if (i < n4) {
    v4f f = ((const v4f*)src)[i];
    ((v4bf*)dst)[i] = __builtin_convertvector(f, v4bf);
  }
}

// ---------------------------------------------------------------------------
// QKV GEMM (transposed): Y^T = Wqkv * X^T, ping-pong pipelined over K (x2).
// Block: 64 tokens x 64 output channels; wave w -> 16 tokens, 4 channel accs.
// Writes directly into attention layouts:
//   Q[bh,n,64] (pre-scaled by SCALE*log2e), K[bh,n,64], Vt[bh,64,n]
// ---------------------------------------------------------------------------
__global__ __launch_bounds__(128)
void qkv_gemm_kernel(const bf16_t* __restrict__ xb,   // [8192,768]
                     const bf16_t* __restrict__ wb,   // [2304,768]
                     bf16_t* __restrict__ Qb,         // [96,1024,64]
                     bf16_t* __restrict__ Kb,         // [96,1024,64]
                     bf16_t* __restrict__ Vt) {       // [96,64,1024]
  const int wave = threadIdx.x >> 5;
  const int lane = threadIdx.x & 31;
  const int tb = blockIdx.x * 64 + wave * 16;   // token base of this wave
  const int cb = blockIdx.y * 64;               // output-channel base (of 2304)

  const bf16_t* xrow = xb + (size_t)tb * DIMC;
  const bf16_t* wrow = wb + (size_t)cb * DIMC;

  v8f acc[4] = {};
  v16bf b0, b1, a0[4], a1[4];

  // prologue: k = 0 into ping
  b0 = load_ab_frag(xrow, DIMC);
#pragma unroll
  for (int ci = 0; ci < 4; ++ci)
    a0[ci] = load_ab_frag(wrow + (size_t)(16 * ci) * DIMC, DIMC);

  // steady state: branch-free body, ping-pong registers
  for (int k0 = 0; k0 < DIMC - 64; k0 += 64) {
    b1 = load_ab_frag(xrow + k0 + 32, DIMC);
#pragma unroll
    for (int ci = 0; ci < 4; ++ci)
      a1[ci] = load_ab_frag(wrow + (size_t)(16 * ci) * DIMC + k0 + 32, DIMC);
#pragma unroll
    for (int ci = 0; ci < 4; ++ci)
      acc[ci] = wmma_bf16(a0[ci], b0, acc[ci]);

    b0 = load_ab_frag(xrow + k0 + 64, DIMC);
#pragma unroll
    for (int ci = 0; ci < 4; ++ci)
      a0[ci] = load_ab_frag(wrow + (size_t)(16 * ci) * DIMC + k0 + 64, DIMC);
#pragma unroll
    for (int ci = 0; ci < 4; ++ci)
      acc[ci] = wmma_bf16(a1[ci], b1, acc[ci]);
  }
  // epilogue: k = DIMC-64 in ping, load DIMC-32 into pong
  b1 = load_ab_frag(xrow + DIMC - 32, DIMC);
#pragma unroll
  for (int ci = 0; ci < 4; ++ci)
    a1[ci] = load_ab_frag(wrow + (size_t)(16 * ci) * DIMC + DIMC - 32, DIMC);
#pragma unroll
  for (int ci = 0; ci < 4; ++ci) acc[ci] = wmma_bf16(a0[ci], b0, acc[ci]);
#pragma unroll
  for (int ci = 0; ci < 4; ++ci) acc[ci] = wmma_bf16(a1[ci], b1, acc[ci]);

  // Lane holds token = tb + (lane&15); channel m = 16ci + kh + r within tile.
  const int tok = tb + (lane & 15);
  const int b   = tok >> 10;
  const int n   = tok & 1023;
  const int kh  = (lane & 16) >> 1;
  const int which = cb / DIMC;                  // 0=q 1=k 2=v (64 | 768)
  const int h     = (cb % DIMC) >> 6;
  const size_t bh = (size_t)(b * NHEADS + h);

#pragma unroll
  for (int ci = 0; ci < 4; ++ci) {
    const int d0 = 16 * ci + kh;                // 8 consecutive d values
    if (which == 0) {
      v8f t = acc[ci] * Q_SCALE;
      *(v8bf*)(Qb + (bh * SEQ + n) * HEADD + d0) = __builtin_convertvector(t, v8bf);
    } else if (which == 1) {
      *(v8bf*)(Kb + (bh * SEQ + n) * HEADD + d0) =
          __builtin_convertvector(acc[ci], v8bf);
    } else {
#pragma unroll
      for (int r = 0; r < 8; ++r)
        Vt[(bh * HEADD + d0 + r) * SEQ + n] = f2bf(acc[ci][r]);
    }
  }
}

// ---------------------------------------------------------------------------
// Flash attention per (b,h). Block: 64 queries (wave -> 16), loop 64-key tiles.
//   S^T = K * Q^T        (lane owns one query column -> lane-local softmax)
//   ctx^T = V^T * P^T    (P^T fragments packed straight from S^T accumulators)
// K and V^T tiles staged via async copy into double-buffered LDS, shared by
// all 4 waves. Softmax in base-2. Writes ctx as bf16 [B,N,C].
// ---------------------------------------------------------------------------
__global__ __launch_bounds__(128)
void attn_kernel(const bf16_t* __restrict__ Qb,
                 const bf16_t* __restrict__ Kb,
                 const bf16_t* __restrict__ Vt,
                 bf16_t* __restrict__ ctxb) {         // [8192,768]
  __shared__ bf16_t ldsK[2][64 * 64];   // [key 0..63][d 0..63]
  __shared__ bf16_t ldsV[2][64 * 64];   // [d 0..63][key 0..63]

  const int tid  = threadIdx.x;
  const int wave = tid >> 5;
  const int lane = tid & 31;
  const int bh   = blockIdx.y;                        // 0..95
  const int qb0  = blockIdx.x * 64 + wave * 16;

  const bf16_t* Q = Qb + (size_t)bh * SEQ * HEADD;
  const bf16_t* K = Kb + (size_t)bh * SEQ * HEADD;
  const bf16_t* V = Vt + (size_t)bh * HEADD * SEQ;

  // Q^T B-fragments are invariant over the key loop: d-halves 0..31 / 32..63.
  const v16bf qf0 = load_ab_frag(Q + (size_t)qb0 * HEADD +  0, HEADD);
  const v16bf qf1 = load_ab_frag(Q + (size_t)qb0 * HEADD + 32, HEADD);

  // prologue: stage tile 0
  stage_tile_async(ldsK[0], K, HEADD, tid);
  stage_tile_async(ldsV[0], V, SEQ, tid);

  v8f   ctxacc[4] = {};                 // ctx^T: d-groups of 16, queries = cols
  float m_run = -__builtin_inff();
  float l_run = 0.0f;

  for (int t = 0; t < SEQ / 64; ++t) {
    wait_async_all();          // this wave's copies into buf[t&1] complete
    __syncthreads();           // all waves' copies complete; buf[(t+1)&1] free

    if (t + 1 < SEQ / 64) {    // overlap next tile's copy with this tile's math
      stage_tile_async(ldsK[(t + 1) & 1], K + (size_t)(t + 1) * 64 * HEADD,
                       HEADD, tid);
      stage_tile_async(ldsV[(t + 1) & 1], V + (size_t)(t + 1) * 64, SEQ, tid);
    }

    const bf16_t* Ktile = ldsK[t & 1];
    const bf16_t* Vtile = ldsV[t & 1];

    // ---- S^T = K * Q^T : hoist all 8 K-fragments, then 8 back-to-back WMMAs
    v16bf kf[4][2];
#pragma unroll
    for (int j = 0; j < 4; ++j) {
      kf[j][0] = load_ab_frag(Ktile + 16 * j * 64 +  0, 64);
      kf[j][1] = load_ab_frag(Ktile + 16 * j * 64 + 32, 64);
    }
    v8f s[4];
#pragma unroll
    for (int j = 0; j < 4; ++j) {
      v8f tacc = {};
      tacc = wmma_bf16(kf[j][0], qf0, tacc);
      tacc = wmma_bf16(kf[j][1], qf1, tacc);
      s[j] = tacc;
    }

    // ---- online softmax in base-2 (log2e folded into Q)
    float mt = -__builtin_inff();
#pragma unroll
    for (int j = 0; j < 4; ++j)
#pragma unroll
      for (int r = 0; r < 8; ++r) mt = fmaxf(mt, s[j][r]);
    mt = fmaxf(mt, __shfl_xor(mt, 16, 32));       // partner lane = same query
    const float m_new = fmaxf(m_run, mt);
    const float alpha = exp2f(m_run - m_new);
    float lsum = 0.0f;
#pragma unroll
    for (int j = 0; j < 4; ++j)
#pragma unroll
      for (int r = 0; r < 8; ++r) {
        float p = exp2f(s[j][r] - m_new);
        s[j][r] = p;
        lsum += p;
      }
    l_run = l_run * alpha + lsum;
    m_run = m_new;
#pragma unroll
    for (int i = 0; i < 4; ++i) ctxacc[i] = ctxacc[i] * alpha;

    // ---- P^T B-fragments: packed f32->bf16 (v_cvt_pk_bf16_f32), lane-local
    v8bf c0 = __builtin_convertvector(s[0], v8bf);
    v8bf c1 = __builtin_convertvector(s[1], v8bf);
    v8bf c2 = __builtin_convertvector(s[2], v8bf);
    v8bf c3 = __builtin_convertvector(s[3], v8bf);
    v16bf p01, p23;
#pragma unroll
    for (int e = 0; e < 8; ++e) {
      p01[e] = c0[e]; p01[e + 8] = c1[e];
      p23[e] = c2[e]; p23[e + 8] = c3[e];
    }

    // ---- ctx^T += V^T * P^T : hoist all 8 V-fragments, then 8 WMMAs
    v16bf vf[4][2];
#pragma unroll
    for (int i = 0; i < 4; ++i) {
      vf[i][0] = load_ab_frag(Vtile + 16 * i * 64 +  0, 64);
      vf[i][1] = load_ab_frag(Vtile + 16 * i * 64 + 32, 64);
    }
#pragma unroll
    for (int i = 0; i < 4; ++i) {
      ctxacc[i] = wmma_bf16(vf[i][0], p01, ctxacc[i]);
      ctxacc[i] = wmma_bf16(vf[i][1], p23, ctxacc[i]);
    }
  }

  const float l_tot = l_run + __shfl_xor(l_run, 16, 32);
  const float inv   = 1.0f / l_tot;

  // Lane: query q = qb0+(lane&15); d = 16i + kh + r (8 contiguous per acc)
  const int q  = qb0 + (lane & 15);
  const int kh = (lane & 16) >> 1;
  const int b  = bh / NHEADS;
  const int h  = bh % NHEADS;
  bf16_t* dst = ctxb + ((size_t)(b * SEQ + q)) * DIMC + h * HEADD;
#pragma unroll
  for (int i = 0; i < 4; ++i) {
    v8f t = ctxacc[i] * inv;
    *(v8bf*)(dst + 16 * i + kh) = __builtin_convertvector(t, v8bf);
  }
}

// ---------------------------------------------------------------------------
// Output projection (transposed GEMM, ping-pong x2) + bias, fp32 output.
// ---------------------------------------------------------------------------
__global__ __launch_bounds__(128)
void proj_gemm_kernel(const bf16_t* __restrict__ ctxb,  // [8192,768]
                      const bf16_t* __restrict__ wb,    // [768,768]
                      const float*  __restrict__ bias,  // [768]
                      float* __restrict__ out) {        // [8192,768]
  const int wave = threadIdx.x >> 5;
  const int lane = threadIdx.x & 31;
  const int tb = blockIdx.x * 64 + wave * 16;
  const int cb = blockIdx.y * 64;

  const bf16_t* xrow = ctxb + (size_t)tb * DIMC;
  const bf16_t* wrow = wb + (size_t)cb * DIMC;

  v8f acc[4] = {};
  v16bf b0, b1, a0[4], a1[4];

  b0 = load_ab_frag(xrow, DIMC);
#pragma unroll
  for (int ci = 0; ci < 4; ++ci)
    a0[ci] = load_ab_frag(wrow + (size_t)(16 * ci) * DIMC, DIMC);

  for (int k0 = 0; k0 < DIMC - 64; k0 += 64) {
    b1 = load_ab_frag(xrow + k0 + 32, DIMC);
#pragma unroll
    for (int ci = 0; ci < 4; ++ci)
      a1[ci] = load_ab_frag(wrow + (size_t)(16 * ci) * DIMC + k0 + 32, DIMC);
#pragma unroll
    for (int ci = 0; ci < 4; ++ci)
      acc[ci] = wmma_bf16(a0[ci], b0, acc[ci]);

    b0 = load_ab_frag(xrow + k0 + 64, DIMC);
#pragma unroll
    for (int ci = 0; ci < 4; ++ci)
      a0[ci] = load_ab_frag(wrow + (size_t)(16 * ci) * DIMC + k0 + 64, DIMC);
#pragma unroll
    for (int ci = 0; ci < 4; ++ci)
      acc[ci] = wmma_bf16(a1[ci], b1, acc[ci]);
  }
  b1 = load_ab_frag(xrow + DIMC - 32, DIMC);
#pragma unroll
  for (int ci = 0; ci < 4; ++ci)
    a1[ci] = load_ab_frag(wrow + (size_t)(16 * ci) * DIMC + DIMC - 32, DIMC);
#pragma unroll
  for (int ci = 0; ci < 4; ++ci) acc[ci] = wmma_bf16(a0[ci], b0, acc[ci]);
#pragma unroll
  for (int ci = 0; ci < 4; ++ci) acc[ci] = wmma_bf16(a1[ci], b1, acc[ci]);

  const int tok = tb + (lane & 15);
  const int kh  = (lane & 16) >> 1;
#pragma unroll
  for (int ci = 0; ci < 4; ++ci) {
    const int c0 = cb + 16 * ci + kh;           // 8 contiguous channels
    v4f lo, hi;
#pragma unroll
    for (int r = 0; r < 4; ++r) lo[r] = acc[ci][r]     + bias[c0 + r];
#pragma unroll
    for (int r = 0; r < 4; ++r) hi[r] = acc[ci][r + 4] + bias[c0 + 4 + r];
    float* dst = out + (size_t)tok * DIMC + c0;
    *(v4f*)(dst)     = lo;
    *(v4f*)(dst + 4) = hi;
  }
}

// ---------------------------------------------------------------------------
// Host-side launcher
// ---------------------------------------------------------------------------
extern "C" void kernel_launch(void* const* d_in, const int* in_sizes, int n_in,
                              void* d_out, int out_size, void* d_ws, size_t ws_size,
                              hipStream_t stream) {
  const float* x      = (const float*)d_in[0];   // [8,1024,768]
  const float* w_qkv  = (const float*)d_in[1];   // [2304,768]
  const float* w_proj = (const float*)d_in[2];   // [768,768]
  const float* b_proj = (const float*)d_in[3];   // [768]
  float* out = (float*)d_out;

  // Workspace carve-up (bf16 buffers), 256B aligned.
  char* ws = (char*)d_ws;
  size_t off = 0;
  auto carve = [&](size_t bytes) {
    void* p = ws + off;
    off += (bytes + 255) & ~(size_t)255;
    return p;
  };
  const size_t X_ELEMS   = (size_t)TOKENS * DIMC;      // 6291456
  const size_t WQ_ELEMS  = (size_t)QKV_OUT * DIMC;     // 1769472
  const size_t WP_ELEMS  = (size_t)DIMC * DIMC;        // 589824
  const size_t HEAD_ELMS = (size_t)BATCH * NHEADS * SEQ * HEADD; // 6291456

  bf16_t* xb     = (bf16_t*)carve(X_ELEMS   * 2);
  bf16_t* wqkvb  = (bf16_t*)carve(WQ_ELEMS  * 2);
  bf16_t* wprojb = (bf16_t*)carve(WP_ELEMS  * 2);
  bf16_t* Qb     = (bf16_t*)carve(HEAD_ELMS * 2);
  bf16_t* Kb     = (bf16_t*)carve(HEAD_ELMS * 2);
  bf16_t* Vt     = (bf16_t*)carve(HEAD_ELMS * 2);
  bf16_t* ctxb   = (bf16_t*)carve(X_ELEMS   * 2);
  (void)ws_size; (void)in_sizes; (void)n_in; (void)out_size;

  // 1) fp32 -> bf16 conversions
  {
    int n4 = (int)(X_ELEMS / 4);
    cvt_f32_bf16_kernel<<<(n4 + 255) / 256, 256, 0, stream>>>(x, xb, n4);
    n4 = (int)(WQ_ELEMS / 4);
    cvt_f32_bf16_kernel<<<(n4 + 255) / 256, 256, 0, stream>>>(w_qkv, wqkvb, n4);
    n4 = (int)(WP_ELEMS / 4);
    cvt_f32_bf16_kernel<<<(n4 + 255) / 256, 256, 0, stream>>>(w_proj, wprojb, n4);
  }

  // 2) QKV projection -> Q (scaled), K, V^T
  {
    dim3 grid(TOKENS / 64, QKV_OUT / 64);   // 128 x 36
    qkv_gemm_kernel<<<grid, 128, 0, stream>>>(xb, wqkvb, Qb, Kb, Vt);
  }

  // 3) Flash attention -> ctx (bf16 [B,N,C])
  {
    dim3 grid(SEQ / 64, BATCH * NHEADS);    // 16 x 96
    attn_kernel<<<grid, 128, 0, stream>>>(Qb, Kb, Vt, ctxb);
  }

  // 4) Output projection + bias -> fp32 out
  {
    dim3 grid(TOKENS / 64, DIMC / 64);      // 128 x 12
    proj_gemm_kernel<<<grid, 128, 0, stream>>>(ctxb, wprojb, b_proj, out);
  }
}
